// MoE_35527969472973
// MI455X (gfx1250) — compile-verified, matrix operand order checked by reading, not run on hardware
//
#include <hip/hip_runtime.h>
#include <hip/hip_bf16.h>

// Problem constants (from reference): B=8, S=1024, D=1024, E=8, F=2048, cap=2
#define B_   8
#define S_   1024
#define D_   1024
#define E_   8
#define F_   2048
#define CAP  256     // S * N_EXPERT_CAPACITY / E

typedef __attribute__((ext_vector_type(16))) __bf16 v16bf;
typedef __attribute__((ext_vector_type(8)))  __bf16 bf16x8;
typedef __attribute__((ext_vector_type(8)))  float  v8f;

#if __has_builtin(__builtin_amdgcn_tensor_load_to_lds)
#define USE_TDM 1
#else
#define USE_TDM 0
#endif

#if USE_TDM
typedef unsigned int u32;
typedef u32 __attribute__((ext_vector_type(4))) u32x4;
typedef int  __attribute__((ext_vector_type(8))) i32x8;
typedef int  __attribute__((ext_vector_type(4))) i32x4;

// Build a D# (ISA 08_async_tensor §8) for a 2D bf16 tile load:
//   tile_d0 elements per row, tile_d1 rows, row stride stride0 (elements).
// group0: [1:0]=count=1, [63:32]=lds_addr, [120:64]=global_addr, [127:126]=type=2
// group1: [17:16]=data_size=1 (2B), [79:48]=tensor_dim0, [111:80]=tensor_dim1,
//         [127:112]=tile_dim0, [143:128]=tile_dim1, [207:160]=tensor_dim0_stride
__device__ __forceinline__ void tdm_load_tile_bf16(
    unsigned lds_addr, const void* gptr, unsigned tile_d0, unsigned tile_d1,
    unsigned long long stride0) {
  const unsigned long long ga = (unsigned long long)(uintptr_t)gptr;
  const unsigned long long dim0 = 1ull << 20;  // generous OOB bounds
  const unsigned long long dim1 = 1ull << 20;
  u32x4 g0;
  g0[0] = 1u;                                            // count=1, user mode
  g0[1] = lds_addr;                                      // LDS byte address
  g0[2] = (u32)ga;                                       // global addr low
  g0[3] = (u32)((ga >> 32) & 0x01FFFFFFu) | (2u << 30);  // addr hi + type=2
  i32x8 g1;
  g1[0] = (int)(1u << 16);                               // data_size=1 -> 2B
  g1[1] = (int)((u32)(dim0 & 0xFFFFu) << 16);            // tensor_dim0 lo16
  g1[2] = (int)(((u32)(dim1 & 0xFFFFu) << 16) | (u32)((dim0 >> 16) & 0xFFFFu));
  g1[3] = (int)(((u32)tile_d0 << 16) | (u32)((dim1 >> 16) & 0xFFFFu));
  g1[4] = (int)((u32)tile_d1 & 0xFFFFu);                 // tile_dim1, dim2=0
  g1[5] = (int)(u32)(stride0 & 0xFFFFFFFFull);           // dim0_stride lo32
  g1[6] = (int)(u32)((stride0 >> 32) & 0xFFFFull);       // dim0_stride hi16
  g1[7] = 0;
  i32x4 z4 = {0, 0, 0, 0};
  i32x8 z8 = {0, 0, 0, 0, 0, 0, 0, 0};
  // 6-arg form (clang-23 / therock-10.0 HIP headers)
  __builtin_amdgcn_tensor_load_to_lds(g0, g1, z4, z4, z8, 0);
}
#endif

// ---------------------------------------------------------------------------
// 1) Router: logits = x @ choice_w^T in fp32 (selection must stay exact-ish),
//    softmax over E, written transposed as probsT[b][e][s].
// ---------------------------------------------------------------------------
__global__ __launch_bounds__(256) void moe_router_kernel(
    const float* __restrict__ x, const float* __restrict__ cw,
    float* __restrict__ probsT) {
  const int token = blockIdx.x;
  const int b = token >> 10;
  const int s = token & (S_ - 1);
  const float* xr = x + (size_t)token * D_;

  float acc[E_];
#pragma unroll
  for (int e = 0; e < E_; ++e) acc[e] = 0.0f;

  for (int d = threadIdx.x; d < D_; d += 256) {
    const float xv = xr[d];
#pragma unroll
    for (int e = 0; e < E_; ++e) acc[e] = fmaf(xv, cw[e * D_ + d], acc[e]);
  }

#pragma unroll
  for (int off = 16; off > 0; off >>= 1) {
#pragma unroll
    for (int e = 0; e < E_; ++e) acc[e] += __shfl_xor(acc[e], off, 32);
  }
  __shared__ float wsum[8][E_];
  const int wave = threadIdx.x >> 5;
  const int lane = threadIdx.x & 31;
  if (lane == 0) {
#pragma unroll
    for (int e = 0; e < E_; ++e) wsum[wave][e] = acc[e];
  }
  __syncthreads();
  if (threadIdx.x == 0) {
    float l[E_];
#pragma unroll
    for (int e = 0; e < E_; ++e) {
      float t = 0.0f;
#pragma unroll
      for (int w = 0; w < 8; ++w) t += wsum[w][e];
      l[e] = t;
    }
    float mx = l[0];
#pragma unroll
    for (int e = 1; e < E_; ++e) mx = fmaxf(mx, l[e]);
    float ex[E_], denom = 0.0f;
#pragma unroll
    for (int e = 0; e < E_; ++e) { ex[e] = __expf(l[e] - mx); denom += ex[e]; }
    const float inv = 1.0f / denom;
#pragma unroll
    for (int e = 0; e < E_; ++e)
      probsT[((size_t)(b * E_ + e)) * S_ + s] = ex[e] * inv;
  }
}

// ---------------------------------------------------------------------------
// 2) Top-k per (b,e): bitonic sort of 1024 (value,index) pairs in LDS.
// ---------------------------------------------------------------------------
__global__ __launch_bounds__(256) void moe_topk_kernel(
    const float* __restrict__ probsT, float* __restrict__ G,
    int* __restrict__ I) {
  __shared__ float v[S_];
  __shared__ int   ix[S_];
  const int be = blockIdx.x;
  const float* p = probsT + (size_t)be * S_;

  for (int i = threadIdx.x; i < S_; i += 256) { v[i] = p[i]; ix[i] = i; }
  __syncthreads();

  for (int size = 2; size <= S_; size <<= 1) {
    for (int stride = size >> 1; stride > 0; stride >>= 1) {
      for (int t = threadIdx.x; t < (S_ / 2); t += 256) {
        const int i = ((t / stride) * (stride << 1)) + (t & (stride - 1));
        const int j = i + stride;
        const bool dir_desc = ((i & size) == 0);
        const float va = v[i], vb = v[j];
        const int ia = ix[i], ib = ix[j];
        const bool a_first = (va > vb) || (va == vb && ia < ib);
        if (a_first != dir_desc) {
          v[i] = vb; v[j] = va; ix[i] = ib; ix[j] = ia;
        }
      }
      __syncthreads();
    }
  }
  for (int i = threadIdx.x; i < CAP; i += 256) {
    G[(size_t)be * CAP + i] = v[i];
    I[(size_t)be * CAP + i] = ix[i];
  }
}

// ---------------------------------------------------------------------------
// WMMA tiling: 128(M) x 128(N) per block, K-step 32, double-buffered LDS.
// 8 waves in 2(M)x4(N); each wave owns 64x32 = 4x2 accumulators.
// A fragment: lane M=lane%16, kb=8*(lane>=16); elems 0..7 -> K=kb+j,
// 8..15 -> K=16+kb+(j-8): two contiguous 16B LDS loads. B mirrored (Bs[n][k]).
// C/D: VGPR r -> M=r+8*(lane>=16), N=lane%16.
// ---------------------------------------------------------------------------
#define MT 128
#define NT 128
#define KT 32

__global__ __launch_bounds__(256) void moe_gemm1_silu_kernel(
    const float* __restrict__ x, const float* __restrict__ w1,
    const int* __restrict__ I, __bf16* __restrict__ h) {
  __shared__ __bf16 As[2][MT][KT];
  __shared__ __bf16 Bs[2][NT][KT];

  const int be = blockIdx.z;
  const int b  = be >> 3;
  const int e  = be & 7;
  const int m0 = blockIdx.y * MT;
  const int n0 = blockIdx.x * NT;
  const int tid  = threadIdx.x;
  const int wave = tid >> 5;
  const int lane = tid & 31;
  const int wm = (wave & 1) * 64;
  const int wn = (wave >> 1) * 32;

  // A loader: 2 threads per gathered token row, 16 floats each
  const int a_row = tid >> 1;
  const int a_col = (tid & 1) * 16;
  const int tokA  = I[be * CAP + m0 + a_row];
  const float* aptr = x + ((size_t)b * S_ + tokA) * D_ + a_col;
  // B loader: 4 k-rows x 4 n-cols per thread, transposed into Bs[n][k]
  const int b_k = wave << 2;
  const int b_n = (tid & 31) << 2;
  const float* bptr = w1 + (size_t)e * D_ * F_ + n0 + b_n;

  auto load_tiles = [&](int kk, int bufi) {
    float4 av[4], bv[4];
#pragma unroll
    for (int i = 0; i < 4; ++i) av[i] = *(const float4*)(aptr + kk + i * 4);
#pragma unroll
    for (int i = 0; i < 4; ++i)
      bv[i] = *(const float4*)(bptr + (size_t)(kk + b_k + i) * F_);
#pragma unroll
    for (int i = 0; i < 4; ++i) {
      As[bufi][a_row][a_col + i * 4 + 0] = (__bf16)av[i].x;
      As[bufi][a_row][a_col + i * 4 + 1] = (__bf16)av[i].y;
      As[bufi][a_row][a_col + i * 4 + 2] = (__bf16)av[i].z;
      As[bufi][a_row][a_col + i * 4 + 3] = (__bf16)av[i].w;
    }
#pragma unroll
    for (int i = 0; i < 4; ++i) {
      Bs[bufi][b_n + 0][b_k + i] = (__bf16)bv[i].x;
      Bs[bufi][b_n + 1][b_k + i] = (__bf16)bv[i].y;
      Bs[bufi][b_n + 2][b_k + i] = (__bf16)bv[i].z;
      Bs[bufi][b_n + 3][b_k + i] = (__bf16)bv[i].w;
    }
  };

  v8f acc[4][2] = {};
  const int mrow = lane & 15;
  const int kb   = (lane >> 4) << 3;

  load_tiles(0, 0);
  __syncthreads();
  int buf = 0;
  for (int k0 = 0; k0 < D_; k0 += KT) {
    if (k0 + KT < D_) {
      __builtin_prefetch(aptr + k0 + 2 * KT, 0, 1);
      load_tiles(k0 + KT, buf ^ 1);
    }
    v16bf af[4], bfr[2];
#pragma unroll
    for (int i = 0; i < 4; ++i) {
      const __bf16* p = &As[buf][wm + i * 16 + mrow][kb];
      ((bf16x8*)&af[i])[0] = *(const bf16x8*)p;
      ((bf16x8*)&af[i])[1] = *(const bf16x8*)(p + 16);
    }
#pragma unroll
    for (int j = 0; j < 2; ++j) {
      const __bf16* p = &Bs[buf][wn + j * 16 + mrow][kb];
      ((bf16x8*)&bfr[j])[0] = *(const bf16x8*)p;
      ((bf16x8*)&bfr[j])[1] = *(const bf16x8*)(p + 16);
    }
#pragma unroll
    for (int i = 0; i < 4; ++i) {
#pragma unroll
      for (int j = 0; j < 2; ++j) {
        acc[i][j] = __builtin_amdgcn_wmma_f32_16x16x32_bf16(
            false, af[i], false, bfr[j], (short)0, acc[i][j], false, false);
      }
    }
    __syncthreads();
    buf ^= 1;
  }

  // epilogue: SiLU via v_rcp_f32, store h as bf16
  const int mh = (lane >> 4) << 3;
#pragma unroll
  for (int i = 0; i < 4; ++i) {
#pragma unroll
    for (int j = 0; j < 2; ++j) {
#pragma unroll
      for (int r = 0; r < 8; ++r) {
        const int m = m0 + wm + i * 16 + r + mh;
        const int n = n0 + wn + j * 16 + mrow;
        const float vv = acc[i][j][r];
        const float sv = vv * __builtin_amdgcn_rcpf(1.0f + __expf(-vv));
        h[((size_t)be * CAP + m) * F_ + n] = (__bf16)sv;
      }
    }
  }
}

__global__ __launch_bounds__(256) void moe_gemm2_scatter_kernel(
    const __bf16* __restrict__ h, const float* __restrict__ w2,
    const int* __restrict__ I, const float* __restrict__ G,
    float* __restrict__ out) {
  __shared__ __bf16 As[2][MT][KT];
  __shared__ __bf16 Bs[2][NT][KT];

  const int be = blockIdx.z;
  const int b  = be >> 3;
  const int e  = be & 7;
  const int m0 = blockIdx.y * MT;
  const int n0 = blockIdx.x * NT;
  const int tid  = threadIdx.x;
  const int wave = tid >> 5;
  const int lane = tid & 31;
  const int wm = (wave & 1) * 64;
  const int wn = (wave >> 1) * 32;

#if !USE_TDM
  const int a_row = tid >> 1;
  const int a_col = (tid & 1) * 16;
  const __bf16* aptr = h + ((size_t)be * CAP + m0 + a_row) * F_ + a_col;
#endif
  const int b_k = wave << 2;
  const int b_n = (tid & 31) << 2;
  const float* bptr = w2 + (size_t)e * F_ * D_ + n0 + b_n;

  auto load_tiles = [&](int kk, int bufi) {
#if USE_TDM
    // Async tensor DMA: h tile (MT rows x KT bf16, row stride F_) -> LDS.
    if (tid < 32) {
      tdm_load_tile_bf16((unsigned)(uintptr_t)&As[bufi][0][0],
                         h + ((size_t)be * CAP + m0) * F_ + kk,
                         KT, MT, (unsigned long long)F_);
    }
#else
    const bf16x8* src = (const bf16x8*)(aptr + kk);
    const bf16x8 v0 = src[0];
    const bf16x8 v1 = src[1];
    *(bf16x8*)&As[bufi][a_row][a_col + 0] = v0;
    *(bf16x8*)&As[bufi][a_row][a_col + 8] = v1;
#endif
    float4 bv[4];
#pragma unroll
    for (int i = 0; i < 4; ++i)
      bv[i] = *(const float4*)(bptr + (size_t)(kk + b_k + i) * D_);
#pragma unroll
    for (int i = 0; i < 4; ++i) {
      Bs[bufi][b_n + 0][b_k + i] = (__bf16)bv[i].x;
      Bs[bufi][b_n + 1][b_k + i] = (__bf16)bv[i].y;
      Bs[bufi][b_n + 2][b_k + i] = (__bf16)bv[i].z;
      Bs[bufi][b_n + 3][b_k + i] = (__bf16)bv[i].w;
    }
  };

  v8f acc[4][2] = {};
  const int mrow = lane & 15;
  const int kb   = (lane >> 4) << 3;

  load_tiles(0, 0);
#if USE_TDM
  if (tid < 32) __builtin_amdgcn_s_wait_tensorcnt(0);
#endif
  __syncthreads();
  int buf = 0;
  for (int k0 = 0; k0 < F_; k0 += KT) {
    const bool has_next = (k0 + KT) < F_;
    if (has_next) {
      __builtin_prefetch(bptr + (size_t)(k0 + 2 * KT) * D_, 0, 1);
      load_tiles(k0 + KT, buf ^ 1);
    }
    v16bf af[4], bfr[2];
#pragma unroll
    for (int i = 0; i < 4; ++i) {
      const __bf16* p = &As[buf][wm + i * 16 + mrow][kb];
      ((bf16x8*)&af[i])[0] = *(const bf16x8*)p;
      ((bf16x8*)&af[i])[1] = *(const bf16x8*)(p + 16);
    }
#pragma unroll
    for (int j = 0; j < 2; ++j) {
      const __bf16* p = &Bs[buf][wn + j * 16 + mrow][kb];
      ((bf16x8*)&bfr[j])[0] = *(const bf16x8*)p;
      ((bf16x8*)&bfr[j])[1] = *(const bf16x8*)(p + 16);
    }
#pragma unroll
    for (int i = 0; i < 4; ++i) {
#pragma unroll
      for (int j = 0; j < 2; ++j) {
        acc[i][j] = __builtin_amdgcn_wmma_f32_16x16x32_bf16(
            false, af[i], false, bfr[j], (short)0, acc[i][j], false, false);
      }
    }
#if USE_TDM
    if (has_next && tid < 32) __builtin_amdgcn_s_wait_tensorcnt(0);
#endif
    __syncthreads();
    buf ^= 1;
  }

  // epilogue: gate and scatter-add back to token positions
  const int mh = (lane >> 4) << 3;
#pragma unroll
  for (int i = 0; i < 4; ++i) {
#pragma unroll
    for (int r = 0; r < 8; ++r) {
      const int m   = m0 + wm + i * 16 + r + mh;
      const int tok = I[be * CAP + m];
      const float g = G[be * CAP + m];
      float* orow = out + ((size_t)b * S_ + tok) * D_ + n0;
#pragma unroll
      for (int j = 0; j < 2; ++j) {
        atomicAdd(orow + wn + j * 16 + mrow, g * acc[i][j][r]);
      }
    }
  }
}

// ---------------------------------------------------------------------------
// Launch
// ---------------------------------------------------------------------------
extern "C" void kernel_launch(void* const* d_in, const int* in_sizes, int n_in,
                              void* d_out, int out_size, void* d_ws,
                              size_t ws_size, hipStream_t stream) {
  const float* x  = (const float*)d_in[0];
  const float* cw = (const float*)d_in[1];
  const float* w1 = (const float*)d_in[2];
  const float* w2 = (const float*)d_in[3];
  float* out = (float*)d_out;

  char* ws = (char*)d_ws;
  float* probsT = (float*)ws;  ws += (size_t)B_ * E_ * S_ * sizeof(float);
  float* G      = (float*)ws;  ws += (size_t)B_ * E_ * CAP * sizeof(float);
  int*   I      = (int*)ws;    ws += (size_t)B_ * E_ * CAP * sizeof(int);
  __bf16* h     = (__bf16*)ws; // B*E*CAP*F bf16 = 64 MiB

  (void)hipMemsetAsync(d_out, 0, (size_t)out_size * sizeof(float), stream);

  moe_router_kernel<<<B_ * S_, 256, 0, stream>>>(x, cw, probsT);
  moe_topk_kernel<<<B_ * E_, 256, 0, stream>>>(probsT, G, I);

  dim3 g1(F_ / NT, CAP / MT, B_ * E_);
  moe_gemm1_silu_kernel<<<g1, 256, 0, stream>>>(x, w1, I, h);

  dim3 g2(D_ / NT, CAP / MT, B_ * E_);
  moe_gemm2_scatter_kernel<<<g2, 256, 0, stream>>>(h, w2, I, G, out);
}
